// Attention_82386062671980
// MI455X (gfx1250) — matrix operand order repaired and checked by reading, not compiled, not run on hardware
//
#include <hip/hip_runtime.h>
#include <hip/hip_bf16.h>

// ---------------- problem constants ----------------
#define NUM_HEADS 32
#define HEAD_DIM  128
#define HIDDEN    4096      // NUM_HEADS * HEAD_DIM
#define BATCH     2
#define SEQ       2048
#define MROWS     (BATCH * SEQ)   // 4096 combined batch*seq rows

typedef __bf16 bf16;
typedef __attribute__((ext_vector_type(16))) __bf16 v16bf;
typedef __attribute__((ext_vector_type(8)))  float  v8f;
typedef __attribute__((ext_vector_type(4)))  unsigned v4u;
typedef __attribute__((ext_vector_type(8)))  unsigned v8u;

// ---------------- CDNA5 async global->LDS copy (ASYNCcnt-tracked) ----------
// LDS generic pointers have the LDS byte offset in the low 32 bits.
__device__ __forceinline__ void async_copy_b128(void* lds_dst, const void* gsrc) {
  unsigned lds = (unsigned)(unsigned long long)lds_dst;
  asm volatile("global_load_async_to_lds_b128 %0, %1, off"
               :: "v"(lds), "v"(gsrc)
               : "memory");
}
__device__ __forceinline__ void wait_async0() {
  asm volatile("s_wait_asynccnt 0" ::: "memory");
}

// ---------------- CDNA5 Tensor Data Mover: 2D tile global->LDS -------------
// Builds a D# (ISA 08_async_tensor §8) and issues TENSOR_LOAD_TO_LDS.
// Wave-level op (EXEC ignored); caller must s_wait_tensorcnt before readers.
__device__ __forceinline__ void tdm_load_2d(void* lds_dst, const void* gsrc,
                                            unsigned tile_d0 /*elems*/,
                                            unsigned tile_d1 /*rows*/,
                                            unsigned row_stride /*elems*/) {
  unsigned long long ga = (unsigned long long)gsrc;
  unsigned lds = (unsigned)(unsigned long long)lds_dst;
  v4u g0;
  g0[0] = 0x1u;                                   // count=1, user descriptor
  g0[1] = lds;                                    // lds_addr
  g0[2] = (unsigned)(ga & 0xffffffffu);           // global_addr[31:0]
  g0[3] = (unsigned)((ga >> 32) & 0x01ffffffu)    // global_addr[56:32]
          | 0x80000000u;                          // type=2 ("image")
  v8u g1;
  unsigned td0 = tile_d0, td1 = tile_d1;          // tensor dims = tile dims
  g1[0] = 0x00010000u;                            // data_size=1 (2 bytes)
  g1[1] = (td0 & 0xffffu) << 16;                  // tensor_dim0[15:0]
  g1[2] = ((td0 >> 16) & 0xffffu) | ((td1 & 0xffffu) << 16); // td0 hi | td1 lo
  g1[3] = ((td1 >> 16) & 0xffffu) | ((tile_d0 & 0xffffu) << 16); // tile_dim0
  g1[4] = (tile_d1 & 0xffffu);                    // tile_dim1 | tile_dim2=0
  g1[5] = row_stride;                             // tensor_dim0_stride[31:0]
  g1[6] = 0u;                                     // stride0 hi | stride1 lo
  g1[7] = 0u;
  asm volatile("tensor_load_to_lds %0, %1" :: "s"(g0), "s"(g1) : "memory");
}

// ---------------- WMMA fragment helpers (ISA 7.12.2 layouts) ----------------
// A (16x32 bf16): lane L holds row M=L%16. half = L/16.
//   VGPR v (v<4): K = half*8 + 2v, 2v+1 ; VGPR v (v>=4): K = 16 + half*8 + 2(v-4), +1
__device__ __forceinline__ v16bf load_a_frag_row(const bf16* rowp, int half) {
  union { v16bf v; unsigned u[8]; } r;
#pragma unroll
  for (int i = 0; i < 8; ++i) {
    int k = (i < 4) ? (half * 8 + 2 * i) : (16 + half * 8 + 2 * (i - 4));
    r.u[i] = *(const unsigned*)(rowp + k);
  }
  return r.v;
}

// B (32x16 bf16): lane L holds col N=L%16; same K order as A: VGPR v (v<4): K = half*8 + 2v, +1; v>=4: K = 16 + half*8 + 2(v-4), +1.
// colp points at element (K=0) of column N in a K-contiguous layout.
__device__ __forceinline__ v16bf load_b_frag(const bf16* colp, int half) {
  union { v16bf v; unsigned u[8]; } r;
#pragma unroll
  for (int i = 0; i < 8; ++i) {
    int k = (i < 4) ? (half * 8 + 2 * i) : (16 + half * 8 + 2 * (i - 4));
    r.u[i] = *(const unsigned*)(colp + k);
  }
  return r.v;
}

__device__ __forceinline__ v8f wmma_bf16(v16bf a, v16bf b, v8f c) {
  return __builtin_amdgcn_wmma_f32_16x16x32_bf16(false, a, false, b, (short)0, c,
                                                 false, false);
}

// ---------------- fp32 -> bf16 convert ----------------
__global__ __launch_bounds__(256) void f32_to_bf16(const float* __restrict__ src,
                                                   bf16* __restrict__ dst, size_t n) {
  size_t i = (size_t)blockIdx.x * blockDim.x + threadIdx.x;
  size_t stride = (size_t)gridDim.x * blockDim.x;
  for (; i < n; i += stride) dst[i] = (bf16)src[i];
}

// ---------------- tiled bf16 WMMA GEMM: C[MxN] = A[MxK] * B[KxN] ----------------
// block = 256 threads = 8 waves. Block tile 128x64: wave w owns rows w*16..w*16+15
// and computes FOUR 16x16 N-tiles per K-step reusing one A fragment (4 WMMAs/step).
template <typename OutT>
__global__ __launch_bounds__(256) void gemm_bf16(const bf16* __restrict__ A,
                                                 const bf16* __restrict__ B,
                                                 OutT* __restrict__ C,
                                                 int M, int N, int K) {
  __shared__ bf16 sA[128 * 32];       // [m][k] row-major
  __shared__ bf16 sBT[64 * 34];       // [n][k] transposed, K padded to 34
  const int KP = 34;

  int t = threadIdx.x;
  int lane = t & 31, w = t >> 5;
  int l16 = lane & 15, half = lane >> 4;
  int m0 = blockIdx.x * 128, n0 = blockIdx.y * 64;

  v8f acc[4];
#pragma unroll
  for (int nt = 0; nt < 4; ++nt) { v8f z = {}; acc[nt] = z; }

  for (int k0 = 0; k0 < K; k0 += 32) {
    __syncthreads();
    // A tile 128x32: async global->LDS, 16 bf16 (2 x b128) per thread
    {
      int r = t >> 1, cs = (t & 1) * 16;
      const bf16* src = A + (size_t)(m0 + r) * K + k0 + cs;
      bf16* dst = &sA[r * 32 + cs];
      async_copy_b128(dst, src);
      async_copy_b128(dst + 8, src + 8);
    }
    // B tile 32(K)x64(N): transpose into sBT[n][k] (8 bf16 per thread)
    {
      int kk = t >> 3, ns = (t & 7) * 8;
      bf16 tmp[8];
      *(uint4*)tmp = *(const uint4*)(B + (size_t)(k0 + kk) * N + n0 + ns);
#pragma unroll
      for (int j = 0; j < 8; ++j) sBT[(ns + j) * KP + kk] = tmp[j];
      if (k0 + 32 < K)
        __builtin_prefetch(B + (size_t)(k0 + 32 + kk) * N + n0 + ns, 0, 1);
    }
    wait_async0();
    __syncthreads();

    v16bf a = load_a_frag_row(sA + (w * 16 + l16) * 32, half);
#pragma unroll
    for (int nt = 0; nt < 4; ++nt) {
      v16bf b = load_b_frag(sBT + (nt * 16 + l16) * KP, half);
      acc[nt] = wmma_bf16(a, b, acc[nt]);
    }
  }

  // C layout: VGPR r -> row (half*8 + r), col = l16
  int mrow = m0 + w * 16 + half * 8;
#pragma unroll
  for (int nt = 0; nt < 4; ++nt) {
    int ncol = n0 + nt * 16 + l16;
#pragma unroll
    for (int r = 0; r < 8; ++r) {
      C[(size_t)(mrow + r) * N + ncol] = (OutT)acc[nt][r];
    }
  }
}

// ---------------- flash attention (multi-query: shared K/V), causal + ALiBi ----
// grid = (SEQ/64, NUM_HEADS, BATCH); block = 128 threads = 4 waves.
// wave w handles 16 query rows q0 = qtile*64 + w*16.
__global__ __launch_bounds__(128) void attn_fwd(const bf16* __restrict__ Q,
                                                const bf16* __restrict__ KV,
                                                bf16* __restrict__ Oattn) {
  __shared__ bf16 sK[32 * 128];        // [key][kd] row-major (k-contig for B frag)
  __shared__ bf16 sVT[128 * 34];       // [hd][key] transposed (key-contig, pad 34)
  __shared__ bf16 sP[4][16 * 32];      // per-wave P staging, row-major [row][key]

  int t = threadIdx.x;
  int lane = t & 31, w = t >> 5;
  int l16 = lane & 15, half = lane >> 4;
  int qtile = blockIdx.x, head = blockIdx.y, b = blockIdx.z;
  int q0 = qtile * 64 + w * 16;

  const float scale = 0.088388347648318447f;       // 1/sqrt(128)
  float slope = exp2f(-0.25f * (float)(head + 1)); // ALiBi (nh=32 pow2 case)

  // Q fragments for this wave's 16 rows, K = 0..127 in 4 fragments
  const bf16* qrow =
      Q + (size_t)(b * SEQ + q0 + l16) * HIDDEN + head * HEAD_DIM;
  v16bf aQ[4];
#pragma unroll
  for (int f = 0; f < 4; ++f) aQ[f] = load_a_frag_row(qrow + 32 * f, half);

  float mrow[8], lrow[8];
  v8f acc[8];
#pragma unroll
  for (int r = 0; r < 8; ++r) { mrow[r] = -1e30f; lrow[r] = 0.0f; }
#pragma unroll
  for (int n = 0; n < 8; ++n) { v8f z = {}; acc[n] = z; }

  int kend = qtile * 64 + 64;              // block-uniform causal bound
  for (int kb = 0; kb < kend; kb += 32) {
    __syncthreads();
    // ---- K tile (32 keys x 128) via Tensor Data Mover: one D# from wave 0 ----
    if (w == 0) {
      const bf16* gK = KV + (size_t)(b * SEQ + kb) * (2 * HEAD_DIM);
      tdm_load_2d(&sK[0], gK, /*tile_d0=*/HEAD_DIM, /*tile_d1=*/32,
                  /*row_stride=*/2 * HEAD_DIM);
    }
    // ---- V tile: regular loads, transposed into sVT[hd][key] ----
    {
      int r = t >> 2, seg = t & 3;         // r: key row 0..31, seg: 32-elem segment
      const bf16* kvrow = KV + (size_t)(b * SEQ + kb + r) * (2 * HEAD_DIM);
      bf16 tv[32];
      const uint4* vs = (const uint4*)(kvrow + HEAD_DIM + seg * 32);
      ((uint4*)tv)[0] = vs[0]; ((uint4*)tv)[1] = vs[1];
      ((uint4*)tv)[2] = vs[2]; ((uint4*)tv)[3] = vs[3];
#pragma unroll
      for (int c = 0; c < 32; ++c) sVT[(seg * 32 + c) * 34 + r] = tv[c];
    }
    if (w == 0) __builtin_amdgcn_s_wait_tensorcnt(0);
    __syncthreads();

    // ---- scores: two 16x16 tiles (keys kb..kb+15, kb+16..kb+31) ----
    v8f s0 = {}, s1 = {};
#pragma unroll
    for (int f = 0; f < 4; ++f) {
      v16bf b0 = load_b_frag(sK + (size_t)l16 * 128 + 32 * f, half);
      s0 = wmma_bf16(aQ[f], b0, s0);
      v16bf b1 = load_b_frag(sK + (size_t)(16 + l16) * 128 + 32 * f, half);
      s1 = wmma_bf16(aQ[f], b1, s1);
    }

    // ---- online softmax (rows live across the 16 lanes of each half-wave) ----
    float p0[8], p1[8], sc[8];
#pragma unroll
    for (int r = 0; r < 8; ++r) {
      int qpos = q0 + half * 8 + r;
      int key0 = kb + l16;
      int key1 = kb + 16 + l16;
      float x0 = s0[r] * scale + slope * (float)(key0 - qpos);
      float x1 = s1[r] * scale + slope * (float)(key1 - qpos);
      if (key0 > qpos) x0 = -1e30f;
      if (key1 > qpos) x1 = -1e30f;
      float mx = fmaxf(x0, x1);
#pragma unroll
      for (int off = 1; off < 16; off <<= 1)
        mx = fmaxf(mx, __shfl_xor(mx, off, 32));
      float mnew = fmaxf(mrow[r], mx);
      float scr = __expf(mrow[r] - mnew);
      float e0 = __expf(x0 - mnew);
      float e1 = __expf(x1 - mnew);
      float ls = e0 + e1;
#pragma unroll
      for (int off = 1; off < 16; off <<= 1) ls += __shfl_xor(ls, off, 32);
      lrow[r] = lrow[r] * scr + ls;
      mrow[r] = mnew;
      sc[r] = scr; p0[r] = e0; p1[r] = e1;
    }
#pragma unroll
    for (int n = 0; n < 8; ++n) {
#pragma unroll
      for (int r = 0; r < 8; ++r) acc[n][r] *= sc[r];
    }

    // ---- stage P through LDS to convert C-layout -> A-fragment layout ----
    bf16* myP = &sP[w][0];
#pragma unroll
    for (int r = 0; r < 8; ++r) {
      int row = half * 8 + r;
      myP[row * 32 + l16]      = (bf16)p0[r];
      myP[row * 32 + 16 + l16] = (bf16)p1[r];
    }
    asm volatile("s_wait_dscnt 0" ::: "memory");
    v16bf aP = load_a_frag_row(myP + l16 * 32, half);

    // ---- O += P * V : 8 column tiles of 16 over head_dim ----
#pragma unroll
    for (int n = 0; n < 8; ++n) {
      v16bf bV = load_b_frag(sVT + (size_t)(n * 16 + l16) * 34, half);
      acc[n] = wmma_bf16(aP, bV, acc[n]);
    }
  }

  // ---- epilogue: normalize and store bf16 ----
#pragma unroll
  for (int r = 0; r < 8; ++r) {
    float inv = (lrow[r] > 0.0f) ? (1.0f / lrow[r]) : 0.0f;
    size_t row = (size_t)(b * SEQ + q0 + half * 8 + r);
#pragma unroll
    for (int n = 0; n < 8; ++n) {
      Oattn[row * HIDDEN + head * HEAD_DIM + n * 16 + l16] =
          (bf16)(acc[n][r] * inv);
    }
  }
}

// ---------------- host orchestration ----------------
extern "C" void kernel_launch(void* const* d_in, const int* in_sizes, int n_in,
                              void* d_out, int out_size, void* d_ws, size_t ws_size,
                              hipStream_t stream) {
  (void)in_sizes; (void)n_in; (void)out_size; (void)ws_size;
  const float* hs  = (const float*)d_in[0];
  const float* Wq  = (const float*)d_in[1];
  const float* Wkv = (const float*)d_in[2];
  const float* Wo  = (const float*)d_in[3];
  float* out = (float*)d_out;

  const size_t N_HS  = (size_t)MROWS * HIDDEN;          // 16,777,216
  const size_t N_WQ  = (size_t)HIDDEN * HIDDEN;
  const size_t N_WKV = (size_t)HIDDEN * (2 * HEAD_DIM);
  const size_t N_WO  = (size_t)HIDDEN * HIDDEN;

  char* ws = (char*)d_ws;
  size_t off = 0;
  auto carve = [&](size_t elems) {
    bf16* p = (bf16*)(ws + off);
    off += ((elems * sizeof(bf16)) + 255) & ~(size_t)255;
    return p;
  };
  bf16* hs_bf   = carve(N_HS);
  bf16* Wq_bf   = carve(N_WQ);
  bf16* Wkv_bf  = carve(N_WKV);
  bf16* Wo_bf   = carve(N_WO);
  bf16* Q_bf    = carve(N_HS);                          // [MROWS][HIDDEN]
  bf16* KV_bf   = carve((size_t)MROWS * 2 * HEAD_DIM);  // [MROWS][256]
  bf16* attn_bf = carve(N_HS);                          // [MROWS][HIDDEN]

  // fp32 -> bf16 conversions
  f32_to_bf16<<<2048, 256, 0, stream>>>(hs,  hs_bf,  N_HS);
  f32_to_bf16<<<2048, 256, 0, stream>>>(Wq,  Wq_bf,  N_WQ);
  f32_to_bf16<<<512,  256, 0, stream>>>(Wkv, Wkv_bf, N_WKV);
  f32_to_bf16<<<2048, 256, 0, stream>>>(Wo,  Wo_bf,  N_WO);

  // Q = hs @ Wq   (4096 x 4096 x 4096) -> bf16
  gemm_bf16<bf16><<<dim3(MROWS / 128, HIDDEN / 64), 256, 0, stream>>>(
      hs_bf, Wq_bf, Q_bf, MROWS, HIDDEN, HIDDEN);
  // KV = hs @ Wkv (4096 x 256 x 4096) -> bf16
  gemm_bf16<bf16><<<dim3(MROWS / 128, (2 * HEAD_DIM) / 64), 256, 0, stream>>>(
      hs_bf, Wkv_bf, KV_bf, MROWS, 2 * HEAD_DIM, HIDDEN);

  // attention
  attn_fwd<<<dim3(SEQ / 64, NUM_HEADS, BATCH), 128, 0, stream>>>(Q_bf, KV_bf,
                                                                 attn_bf);

  // out = attn @ Wo (4096 x 4096 x 4096) -> fp32
  gemm_bf16<float><<<dim3(MROWS / 128, HIDDEN / 64), 256, 0, stream>>>(
      attn_bf, Wo_bf, out, MROWS, HIDDEN, HIDDEN);
}
